// Qwen2DyMemDecoderLayer_61048665145358
// MI455X (gfx1250) — matrix operand order repaired and checked by reading, not compiled, not run on hardware
//
#include <hip/hip_runtime.h>
#include <cstddef>
#include <cstdint>

// ---------------------------------------------------------------------------
// CDNA5 / gfx1250 decoder layer. All heavy GEMMs use v_wmma_f32_16x16x32_bf16
// (wave32, fp32 accum) with LDS-staged operand panels: the A panel is moved by
// the Tensor Data Mover (tensor_load_to_lds + s_wait_tensorcnt), the B panel is
// transpose-staged by the VALU so the WMMA inner loop reads only contiguous
// LDS. Flash attention re-layouts the probability tile through LDS
// (s_wait_dscnt) between the QK^T and PV WMMAs.
// ---------------------------------------------------------------------------

typedef __bf16 bf16_t;
typedef __attribute__((ext_vector_type(16))) __bf16 v16bf;
typedef __attribute__((ext_vector_type(8)))  float  v8f;
typedef __attribute__((ext_vector_type(8)))  unsigned short us8;
typedef __attribute__((ext_vector_type(4)))  unsigned int v4u;
typedef __attribute__((ext_vector_type(8)))  int v8i;
typedef __attribute__((ext_vector_type(4)))  int v4i;

constexpr int BB = 2, TT = 4096, HDIM = 2048, NHEAD = 16, KVHEAD = 8, DHEAD = 128;
constexpr int SINKN = 128, WINN = 2048, MMEM = 64, GREP = 2, NSTATE = 128;
constexpr int DIN = 2048, KCONV = 4;
constexpr int CONVD = DIN + 2 * GREP * NSTATE;   // 2560
constexpr int FFD = 8192;
constexpr int EEV = TT - WINN - SINKN;           // 1920
constexpr int TT2 = SINKN + MMEM + WINN;         // 2240
constexpr int ZDTD = DIN + CONVD + NHEAD;        // 4624
constexpr int HPG = NHEAD / GREP;                // 8 mamba heads per B/C group

__device__ __forceinline__ unsigned short f2bf(float f) {
  unsigned int u = __float_as_uint(f);
  return (unsigned short)((u + 0x7FFFu + ((u >> 16) & 1u)) >> 16);
}
__device__ __forceinline__ float siluf(float x) { return x / (1.0f + __expf(-x)); }

__device__ __forceinline__ v8f wmma_bf16(v16bf a, v16bf b, v8f c) {
  return __builtin_amdgcn_wmma_f32_16x16x32_bf16(false, a, false, b, (short)0, c,
                                                 false, false);
}

// A-matrix fragment: contiguous 32 K-values at p; layout K = hl*16+i
__device__ __forceinline__ v16bf load_frag_a(const bf16_t* __restrict__ p, int hl) {
  v16bf a;
#pragma unroll
  for (int i = 0; i < 16; i++) a[i] = p[hl * 16 + i];
  return a;
}
// B-matrix fragment where the lane's column is contiguous in memory (transposed LDS)
__device__ __forceinline__ v16bf load_frag_bc(const bf16_t* __restrict__ p, int hl) {
  v16bf a;
#pragma unroll
  for (int i = 0; i < 16; i++) a[i] = p[hl * 16 + i];
  return a;
}
// B-matrix fragment from row-major (K x N): lane holds column, stride N (global)
__device__ __forceinline__ v16bf load_frag_b(const bf16_t* __restrict__ p, size_t stride,
                                             int hl) {
  v16bf a;
#pragma unroll
  for (int i = 0; i < 16; i++) a[i] = p[(size_t)(hl * 16 + i) * stride];
  return a;
}

// ---- Tensor Data Mover: DMA a (rows=64 x cols=32) bf16 tile, row stride Kd
// elements, from global into LDS (packed row-major, matching Ash layout). ----
#if __has_builtin(__builtin_amdgcn_tensor_load_to_lds)
#define HAVE_TDM 1
__device__ __forceinline__ void tdm_load_panel_a(const unsigned short* gptr,
                                                 unsigned lds_off, int Kd, int Mr) {
  unsigned long long ga = (unsigned long long)(uintptr_t)gptr;
  v4u g0;
  g0[0] = 1u;                                   // count=1, user descriptor
  g0[1] = lds_off;                              // lds_addr (bytes)
  g0[2] = (unsigned)(ga & 0xFFFFFFFFu);         // global_addr[31:0]
  g0[3] = (unsigned)((ga >> 32) & 0x01FFFFFFu)  // global_addr[56:32]
          | (2u << 30);                         // type = 2 ("image")
  v8i g1;
  g1[0] = (int)(1u << 16);                      // data_size = 1 -> 2 bytes
  g1[1] = (int)(((unsigned)Kd & 0xFFFFu) << 16);            // tensor_dim0 lo16
  g1[2] = (int)((((unsigned)Kd >> 16) & 0xFFFFu)            // tensor_dim0 hi16
                | (((unsigned)Mr & 0xFFFFu) << 16));        // tensor_dim1 lo16
  g1[3] = (int)((((unsigned)Mr >> 16) & 0xFFFFu)            // tensor_dim1 hi16
                | (32u << 16));                             // tile_dim0 = 32
  g1[4] = 64;                                   // tile_dim1 = 64, tile_dim2 = 0
  g1[5] = Kd;                                   // tensor_dim0_stride[31:0]
  g1[6] = 0;                                    // stride hi / dim1_stride lo
  g1[7] = 0;
  v4i gz = {0, 0, 0, 0};
#if defined(__clang_major__) && (__clang_major__ >= 23)
  v8i gz8 = {0, 0, 0, 0, 0, 0, 0, 0};
  __builtin_amdgcn_tensor_load_to_lds(g0, g1, gz, gz, gz8, 0);
#else
  __builtin_amdgcn_tensor_load_to_lds(g0, g1, gz, gz, 0);
#endif
}
#else
#define HAVE_TDM 0
#endif

// ------------------------------ generic WMMA GEMM --------------------------
// C[Mr x Nc] = A[Mr x Kd] (bf16) @ B0[Kd x Nc] (bf16)  (+bias, +addend)
// GATED: silu(A@B0) * (A@B1).  BF16OUT: write ushort-bf16, else f32.
// Block tile 64(M) x 128(N), K-step 32, LDS staged (B transposed in LDS,
// A panel via TDM). Requires: Mr % 64 == 0, Kd % 32 == 0.
template <bool BF16OUT, bool GATED>
__global__ __launch_bounds__(256) void gemm_k(
    const bf16_t* __restrict__ A, const bf16_t* __restrict__ B0,
    const bf16_t* __restrict__ B1, const float* __restrict__ bias,
    const float* __restrict__ addend, float* __restrict__ Cf,
    unsigned short* __restrict__ Cb, int Mr, int Nc, int Kd) {
  __shared__ unsigned short Ash[64 * 32];                      // row-major panel
  __shared__ unsigned short Bsh0[128 * 32];                    // transposed panel
  __shared__ unsigned short Bsh1[(GATED ? 128 : 1) * 32];

  const int tid = threadIdx.x;
  const int wave = tid >> 5, lane = tid & 31;
  const int hl = lane >> 4, lq = lane & 15;
  const int mw = wave >> 2, nw = wave & 3;     // 2 x 4 wave grid -> 32x32 / wave
  const int mblk = blockIdx.y * 64;
  const int nblk = blockIdx.x * 128;

  // staging assignments
  const int ar = tid >> 2;          // 0..63   A row (manual fallback)
  const int ac = (tid & 3) * 8;     // 0..24   A col (8 elems, 16B)
  const int bkr = tid >> 3;         // 0..31   B k-row
  const int bnc = (tid & 7) * 16;   // 0..112  B col base (16 elems)

  const unsigned short* gA = (const unsigned short*)A;
  const unsigned short* gB0 = (const unsigned short*)B0;
  const unsigned short* gB1 = (const unsigned short*)B1;

  v8f acc[2][2] = {{{}, {}}, {{}, {}}};
  v8f gac[2][2] = {{{}, {}}, {{}, {}}};

  for (int k0 = 0; k0 < Kd; k0 += 32) {
    // ---- stage A (64x32): Tensor Data Mover, overlapped with B staging ----
#if HAVE_TDM
    if (wave == 0) {
      tdm_load_panel_a(gA + (size_t)mblk * Kd + k0,
                       (unsigned)(size_t)(void*)&Ash[0], Kd, Mr);
    }
#else
    {
      const unsigned short* gp = gA + (size_t)(mblk + ar) * Kd + k0 + ac;
      *(us8*)&Ash[ar * 32 + ac] = *(const us8*)gp;
    }
#endif
    // ---- stage B (32x128) transposed into LDS ----
    {
      const unsigned short* gp = gB0 + (size_t)(k0 + bkr) * Nc + nblk + bnc;
      __builtin_prefetch(gp + 32 * Nc, 0, 0);  // next K panel -> global_prefetch_b8
      if (nblk + bnc + 15 < Nc) {
        us8 lo = *(const us8*)gp;
        us8 hi = *(const us8*)(gp + 8);
#pragma unroll
        for (int j = 0; j < 8; j++) {
          Bsh0[(bnc + j) * 32 + bkr] = lo[j];
          Bsh0[(bnc + 8 + j) * 32 + bkr] = hi[j];
        }
      } else {
#pragma unroll
        for (int j = 0; j < 16; j++) {
          int col = nblk + bnc + j;
          Bsh0[(bnc + j) * 32 + bkr] = (col < Nc) ? gp[j] : (unsigned short)0;
        }
      }
      if (GATED) {
        const unsigned short* gp1 = gB1 + (size_t)(k0 + bkr) * Nc + nblk + bnc;
        if (nblk + bnc + 15 < Nc) {
          us8 lo = *(const us8*)gp1;
          us8 hi = *(const us8*)(gp1 + 8);
#pragma unroll
          for (int j = 0; j < 8; j++) {
            Bsh1[(bnc + j) * 32 + bkr] = lo[j];
            Bsh1[(bnc + 8 + j) * 32 + bkr] = hi[j];
          }
        } else {
#pragma unroll
          for (int j = 0; j < 16; j++) {
            int col = nblk + bnc + j;
            Bsh1[(bnc + j) * 32 + bkr] = (col < Nc) ? gp1[j] : (unsigned short)0;
          }
        }
      }
    }
#if HAVE_TDM
    if (wave == 0) __builtin_amdgcn_s_wait_tensorcnt(0);  // A panel landed
#endif
    __syncthreads();
    // ---- WMMA on LDS panels: 2 A-frags x 2 B-frags per wave ----
    v16bf a0 = load_frag_a((const bf16_t*)&Ash[(mw * 32 + lq) * 32], hl);
    v16bf a1 = load_frag_a((const bf16_t*)&Ash[(mw * 32 + 16 + lq) * 32], hl);
    v16bf b0f = load_frag_bc((const bf16_t*)&Bsh0[(nw * 32 + lq) * 32], hl);
    v16bf b1f = load_frag_bc((const bf16_t*)&Bsh0[(nw * 32 + 16 + lq) * 32], hl);
    acc[0][0] = wmma_bf16(a0, b0f, acc[0][0]);
    acc[0][1] = wmma_bf16(a0, b1f, acc[0][1]);
    acc[1][0] = wmma_bf16(a1, b0f, acc[1][0]);
    acc[1][1] = wmma_bf16(a1, b1f, acc[1][1]);
    if (GATED) {
      v16bf g0f = load_frag_bc((const bf16_t*)&Bsh1[(nw * 32 + lq) * 32], hl);
      v16bf g1f = load_frag_bc((const bf16_t*)&Bsh1[(nw * 32 + 16 + lq) * 32], hl);
      gac[0][0] = wmma_bf16(a0, g0f, gac[0][0]);
      gac[0][1] = wmma_bf16(a0, g1f, gac[0][1]);
      gac[1][0] = wmma_bf16(a1, g0f, gac[1][0]);
      gac[1][1] = wmma_bf16(a1, g1f, gac[1][1]);
    }
    __syncthreads();
  }
#pragma unroll
  for (int i = 0; i < 2; i++) {
#pragma unroll
    for (int j = 0; j < 2; j++) {
#pragma unroll
      for (int r = 0; r < 8; r++) {
        int row = mblk + mw * 32 + i * 16 + r + 8 * hl;
        int col = nblk + nw * 32 + j * 16 + lq;
        if (col >= Nc) continue;
        size_t idx = (size_t)row * Nc + col;
        float v = acc[i][j][r];
        if (bias) v += bias[col];
        if (addend) v += addend[idx];
        if (GATED) v = siluf(v) * gac[i][j][r];
        if (BF16OUT) Cb[idx] = f2bf(v); else Cf[idx] = v;
      }
    }
  }
}

// ------------------------------ flash attention ----------------------------
// Q: [b,NH,T2,D] bf16, K/V: [b,KVH,T2,D] bf16, O: [b,T2,NH*D] bf16. Causal.
__global__ __launch_bounds__(256) void flash_attn(
    const bf16_t* __restrict__ Q, const bf16_t* __restrict__ Kt,
    const bf16_t* __restrict__ V, unsigned short* __restrict__ Ob) {
  __shared__ unsigned short plds[8][512];  // per-wave 16x32 P tile
  const int wave = threadIdx.x >> 5, lane = threadIdx.x & 31;
  const int hl = lane >> 4, lq = lane & 15;
  const int QT = TT2 / 16;  // 140
  int job = blockIdx.x * 8 + wave;
  int qt = job % QT, bh = job / QT;
  int h = bh % NHEAD, b = bh / NHEAD;
  int hk = h / GREP;
  int q0 = qt * 16;
  const float scale = 0.08838834764831845f;  // 1/sqrt(128)
  const bf16_t* qbase = Q + (size_t)(b * NHEAD + h) * TT2 * DHEAD;
  const bf16_t* kbase = Kt + (size_t)(b * KVHEAD + hk) * TT2 * DHEAD;
  const bf16_t* vbase = V + (size_t)(b * KVHEAD + hk) * TT2 * DHEAD;

  v16bf qfr[4];
#pragma unroll
  for (int kk = 0; kk < 4; kk++)
    qfr[kk] = load_frag_a(qbase + (size_t)(q0 + lq) * DHEAD + kk * 32, hl);

  v8f oacc[8];
#pragma unroll
  for (int nt = 0; nt < 8; nt++) oacc[nt] = (v8f){};
  float mrow[8], lrow[8];
#pragma unroll
  for (int r = 0; r < 8; r++) { mrow[r] = -1e30f; lrow[r] = 0.0f; }

  const int nsteps = (q0 + 16 + 31) / 32;
  for (int kt = 0; kt < nsteps; kt++) {
    int k0 = kt * 32;
    v8f s0 = {}, s1 = {};
#pragma unroll
    for (int kk = 0; kk < 4; kk++) {
      v16bf kf0 = load_frag_bc(kbase + (size_t)(k0 + lq) * DHEAD + kk * 32, hl);
      s0 = wmma_bf16(qfr[kk], kf0, s0);
      v16bf kf1 = load_frag_bc(kbase + (size_t)(k0 + 16 + lq) * DHEAD + kk * 32, hl);
      s1 = wmma_bf16(qfr[kk], kf1, s1);
    }
#pragma unroll
    for (int r = 0; r < 8; r++) {
      int rowi = q0 + r + 8 * hl;
      float e0 = s0[r] * scale, e1 = s1[r] * scale;
      if (k0 + lq > rowi) e0 = -1e30f;
      if (k0 + 16 + lq > rowi) e1 = -1e30f;
      float mx = fmaxf(e0, e1);
#pragma unroll
      for (int d = 1; d < 16; d <<= 1) mx = fmaxf(mx, __shfl_xor(mx, d, 32));
      float mnew = fmaxf(mrow[r], mx);
      float corr = __expf(mrow[r] - mnew);
      float p0 = __expf(e0 - mnew), p1 = __expf(e1 - mnew);
      float ps = p0 + p1;
#pragma unroll
      for (int d = 1; d < 16; d <<= 1) ps += __shfl_xor(ps, d, 32);
      lrow[r] = lrow[r] * corr + ps;
      mrow[r] = mnew;
#pragma unroll
      for (int nt = 0; nt < 8; nt++) oacc[nt][r] *= corr;
      plds[wave][(r + 8 * hl) * 32 + lq] = f2bf(p0);
      plds[wave][(r + 8 * hl) * 32 + 16 + lq] = f2bf(p1);
    }
    asm volatile("s_wait_dscnt 0x0" ::: "memory");
    v16bf pf = load_frag_a((const bf16_t*)&plds[wave][lq * 32], hl);
#pragma unroll
    for (int nt = 0; nt < 8; nt++) {
      v16bf vfr = load_frag_b(vbase + (size_t)k0 * DHEAD + nt * 16 + lq,
                              (size_t)DHEAD, hl);
      oacc[nt] = wmma_bf16(pf, vfr, oacc[nt]);
    }
  }
#pragma unroll
  for (int nt = 0; nt < 8; nt++) {
#pragma unroll
    for (int r = 0; r < 8; r++) {
      int row = q0 + r + 8 * hl;
      float val = oacc[nt][r] / lrow[r];
      Ob[(size_t)(b * TT2 + row) * (NHEAD * DHEAD) + h * DHEAD + nt * 16 + lq] =
          f2bf(val);
    }
  }
}

// ------------------------------ elementwise kernels ------------------------
__global__ void cvt_bf16(const float* __restrict__ s, unsigned short* __restrict__ d,
                         size_t n) {
  size_t i = (size_t)blockIdx.x * 256 + threadIdx.x;
  if (i < n) d[i] = f2bf(s[i]);
}

__global__ void rmsnorm_bf16(const float* __restrict__ X, const float* __restrict__ W,
                             unsigned short* __restrict__ Y, int cols) {
  __shared__ float red[256];
  size_t row = blockIdx.x;
  const float* x = X + row * cols;
  float ss = 0.0f;
  for (int i = threadIdx.x; i < cols; i += 256) { float v = x[i]; ss += v * v; }
  red[threadIdx.x] = ss; __syncthreads();
  for (int s = 128; s > 0; s >>= 1) {
    if ((int)threadIdx.x < s) red[threadIdx.x] += red[threadIdx.x + s];
    __syncthreads();
  }
  float inv = rsqrtf(red[0] / (float)cols + 1e-6f);
  for (int i = threadIdx.x; i < cols; i += 256)
    Y[row * cols + i] = f2bf(x[i] * inv * W[i]);
}

__global__ void gather_hev(const unsigned short* __restrict__ hf,
                           unsigned short* __restrict__ hev) {
  size_t i = (size_t)blockIdx.x * 256 + threadIdx.x;  // BB*EEV*HDIM
  int c = (int)(i % HDIM);
  size_t be = i / HDIM;
  int e = (int)(be % EEV), b = (int)(be / EEV);
  hev[i] = hf[((size_t)(b * TT + SINKN + e)) * HDIM + c];
}

__global__ void mamba_conv(const float* __restrict__ zx, const float* __restrict__ cw,
                           const float* __restrict__ cb, float* __restrict__ xbc) {
  size_t i = (size_t)blockIdx.x * 256 + threadIdx.x;  // BB*EEV*CONVD
  int c = (int)(i % CONVD);
  size_t be = i / CONVD;
  int e = (int)(be % EEV), b = (int)(be / EEV);
  float acc = cb[c];
#pragma unroll
  for (int j = 0; j < KCONV; j++) {
    int ee = e - (KCONV - 1) + j;
    if (ee >= 0)
      acc += zx[((size_t)(b * EEV + ee)) * ZDTD + DIN + c] * cw[c * KCONV + j];
  }
  xbc[i] = siluf(acc);
}

__global__ void dt_decay(const float* __restrict__ zx, const float* __restrict__ dtb,
                         const float* __restrict__ alog, float* __restrict__ dtv,
                         float* __restrict__ dec) {
  size_t i = (size_t)blockIdx.x * 256 + threadIdx.x;  // BB*EEV*NHEAD
  int h = (int)(i % NHEAD);
  size_t row = i / NHEAD;
  float raw = zx[row * ZDTD + DIN + CONVD + h] + dtb[h];
  float sp = (raw > 20.0f) ? raw : log1pf(__expf(raw));
  dtv[i] = sp;
  dec[i] = __expf(-__expf(alog[h]) * sp);
}

__global__ __launch_bounds__(256) void mamba_scan(const float* __restrict__ xbc,
                                                  const float* __restrict__ dtv,
                                                  const float* __restrict__ dec,
                                                  float* __restrict__ ys) {
  __shared__ float xsh[DHEAD], Bsh[NSTATE], Csh[NSTATE];
  int bh = blockIdx.x;
  int h = bh % NHEAD, b = bh / NHEAD;
  int g = h / HPG;
  int tid = threadIdx.x;
  int p = tid >> 1, n0 = (tid & 1) * 64;
  float hs[64];
#pragma unroll
  for (int j = 0; j < 64; j++) hs[j] = 0.0f;
  for (int e = 0; e < EEV; e++) {
    size_t base = ((size_t)(b * EEV + e)) * CONVD;
    if (tid < 128) {
      xsh[tid] = xbc[base + h * DHEAD + tid];
      Bsh[tid] = xbc[base + DIN + g * NSTATE + tid];
    } else {
      Csh[tid - 128] = xbc[base + DIN + GREP * NSTATE + g * NSTATE + (tid - 128)];
    }
    __syncthreads();
    size_t rh = ((size_t)(b * EEV + e)) * NHEAD + h;
    float dcv = dec[rh], dtt = dtv[rh];
    float xc = dtt * xsh[p];
    float acc = 0.0f;
#pragma unroll
    for (int j = 0; j < 64; j++) {
      hs[j] = hs[j] * dcv + xc * Bsh[n0 + j];
      acc += hs[j] * Csh[n0 + j];
    }
    acc += __shfl_xor(acc, 1, 32);
    if ((tid & 1) == 0)
      ys[((size_t)(b * EEV + e)) * DIN + h * DHEAD + p] = acc;
    __syncthreads();
  }
}

__global__ void mem_norm(const float* __restrict__ ys, const float* __restrict__ zx,
                         const float* __restrict__ xbc, const float* __restrict__ mD,
                         const float* __restrict__ nw, const int* __restrict__ midx,
                         unsigned short* __restrict__ out) {
  __shared__ float red[256];
  int r = blockIdx.x;  // BB*MMEM rows
  int m = r % MMEM, b = r / MMEM;
  int e = midx[m];
  size_t row = (size_t)(b * EEV + e);
  float vals[8];
  float ss = 0.0f;
#pragma unroll
  for (int j = 0; j < 8; j++) {
    int i = threadIdx.x + j * 256;
    int h = i >> 7;
    float yv = ys[row * DIN + i] + mD[h] * xbc[row * CONVD + i];
    float z = zx[row * ZDTD + i];
    float v = yv * siluf(z);
    vals[j] = v;
    ss += v * v;
  }
  red[threadIdx.x] = ss; __syncthreads();
  for (int s = 128; s > 0; s >>= 1) {
    if ((int)threadIdx.x < s) red[threadIdx.x] += red[threadIdx.x + s];
    __syncthreads();
  }
  float inv = rsqrtf(red[0] / (float)DIN + 1e-6f);
#pragma unroll
  for (int j = 0; j < 8; j++) {
    int i = threadIdx.x + j * 256;
    out[(size_t)r * DIN + i] = f2bf(vals[j] * inv * nw[i]);
  }
}

__global__ void gather_tokens(const unsigned short* __restrict__ hf,
                              const float* __restrict__ x,
                              unsigned short* __restrict__ h2,
                              float* __restrict__ resid) {
  size_t i = (size_t)blockIdx.x * 256 + threadIdx.x;  // BB*TT2*HDIM
  int c = (int)(i % HDIM);
  size_t bt = i / HDIM;
  int t = (int)(bt % TT2), b = (int)(bt / TT2);
  int src = (t < SINKN + MMEM) ? t : t + (EEV - MMEM);
  size_t si = ((size_t)(b * TT + src)) * HDIM + c;
  h2[i] = hf[si];
  resid[i] = x[si];
}

__global__ void gather_cs(const float* __restrict__ ca, const float* __restrict__ sa,
                          float* __restrict__ c2, float* __restrict__ s2) {
  size_t i = (size_t)blockIdx.x * 256 + threadIdx.x;  // BB*TT2*DHEAD
  int d = (int)(i % DHEAD);
  size_t bt = i / DHEAD;
  int t = (int)(bt % TT2), b = (int)(bt / TT2);
  int src = (t < SINKN + MMEM) ? t : t + (EEV - MMEM);
  size_t si = ((size_t)(b * TT + src)) * DHEAD + d;
  c2[i] = ca[si];
  s2[i] = sa[si];
}

__global__ void rope_q(const float* __restrict__ qf, const float* __restrict__ c2,
                       const float* __restrict__ s2, unsigned short* __restrict__ qb) {
  size_t i = (size_t)blockIdx.x * 256 + threadIdx.x;  // [b,NH,T2,D] order
  int d = (int)(i % DHEAD);
  size_t r = i / DHEAD;
  int t = (int)(r % TT2);
  size_t r2 = r / TT2;
  int h = (int)(r2 % NHEAD), b = (int)(r2 / NHEAD);
  size_t s = ((size_t)(b * TT2 + t)) * (NHEAD * DHEAD) + h * DHEAD;
  float v = qf[s + d];
  float rot = (d < DHEAD / 2) ? -qf[s + d + DHEAD / 2] : qf[s + d - DHEAD / 2];
  size_t ci = ((size_t)(b * TT2 + t)) * DHEAD + d;
  qb[i] = f2bf(v * c2[ci] + rot * s2[ci]);
}

__global__ void assemble_k(const float* __restrict__ kf, const float* __restrict__ mkf,
                           const float* __restrict__ c2, const float* __restrict__ s2,
                           unsigned short* __restrict__ kb) {
  size_t i = (size_t)blockIdx.x * 256 + threadIdx.x;  // [b,KVH,T2,D]
  int d = (int)(i % DHEAD);
  size_t r = i / DHEAD;
  int t = (int)(r % TT2);
  size_t r2 = r / TT2;
  int hk = (int)(r2 % KVHEAD), b = (int)(r2 / KVHEAD);
  float v, rot;
  if (t >= SINKN && t < SINKN + MMEM) {
    size_t s = ((size_t)(b * MMEM + (t - SINKN))) * (KVHEAD * DHEAD) + hk * DHEAD;
    v = mkf[s + d];
    rot = (d < DHEAD / 2) ? -mkf[s + d + DHEAD / 2] : mkf[s + d - DHEAD / 2];
  } else {
    size_t s = ((size_t)(b * TT2 + t)) * (KVHEAD * DHEAD) + hk * DHEAD;
    v = kf[s + d];
    rot = (d < DHEAD / 2) ? -kf[s + d + DHEAD / 2] : kf[s + d - DHEAD / 2];
  }
  size_t ci = ((size_t)(b * TT2 + t)) * DHEAD + d;
  kb[i] = f2bf(v * c2[ci] + rot * s2[ci]);
}

__global__ void assemble_v(const float* __restrict__ vf, const float* __restrict__ mvf,
                           unsigned short* __restrict__ vb) {
  size_t i = (size_t)blockIdx.x * 256 + threadIdx.x;  // [b,KVH,T2,D]
  int d = (int)(i % DHEAD);
  size_t r = i / DHEAD;
  int t = (int)(r % TT2);
  size_t r2 = r / TT2;
  int hk = (int)(r2 % KVHEAD), b = (int)(r2 / KVHEAD);
  float v;
  if (t >= SINKN && t < SINKN + MMEM)
    v = mvf[((size_t)(b * MMEM + (t - SINKN))) * (KVHEAD * DHEAD) + hk * DHEAD + d];
  else
    v = vf[((size_t)(b * TT2 + t)) * (KVHEAD * DHEAD) + hk * DHEAD + d];
  vb[i] = f2bf(v);
}

// ------------------------------ launch -------------------------------------
extern "C" void kernel_launch(void* const* d_in, const int* in_sizes, int n_in,
                              void* d_out, int out_size, void* d_ws, size_t ws_size,
                              hipStream_t stream) {
  (void)in_sizes; (void)n_in; (void)out_size; (void)ws_size;
  const float* x    = (const float*)d_in[0];
  const float* cosA = (const float*)d_in[1];
  const float* sinA = (const float*)d_in[2];
  const int*   midx = (const int*)d_in[3];
  const float* wq = (const float*)d_in[4];
  const float* bq = (const float*)d_in[5];
  const float* wk = (const float*)d_in[6];
  const float* bk = (const float*)d_in[7];
  const float* wv = (const float*)d_in[8];
  const float* bv = (const float*)d_in[9];
  const float* wo = (const float*)d_in[10];
  const float* ln1 = (const float*)d_in[11];
  const float* ln2 = (const float*)d_in[12];
  const float* wg = (const float*)d_in[13];
  const float* wu = (const float*)d_in[14];
  const float* wd = (const float*)d_in[15];
  const float* m_in_w = (const float*)d_in[16];
  const float* m_conv_w = (const float*)d_in[17];
  const float* m_conv_b = (const float*)d_in[18];
  const float* m_dt_bias = (const float*)d_in[19];
  const float* m_A_log = (const float*)d_in[20];
  const float* m_D = (const float*)d_in[21];
  const float* m_norm_w = (const float*)d_in[22];
  const float* m_out_w = (const float*)d_in[23];

  char* wsp = (char*)d_ws;
  size_t off = 0;
  auto alloc = [&](size_t bytes) -> void* {
    void* p = wsp + off;
    off += (bytes + 255) & ~(size_t)255;
    return p;
  };
  auto aB = [&](size_t n) { return (unsigned short*)alloc(n * 2); };
  auto aF = [&](size_t n) { return (float*)alloc(n * 4); };

  unsigned short* Wq = aB((size_t)HDIM * NHEAD * DHEAD);
  unsigned short* Wk = aB((size_t)HDIM * KVHEAD * DHEAD);
  unsigned short* Wv = aB((size_t)HDIM * KVHEAD * DHEAD);
  unsigned short* Wo = aB((size_t)NHEAD * DHEAD * HDIM);
  unsigned short* Wg = aB((size_t)HDIM * FFD);
  unsigned short* Wu = aB((size_t)HDIM * FFD);
  unsigned short* Wd = aB((size_t)FFD * HDIM);
  unsigned short* Wmin = aB((size_t)HDIM * ZDTD);
  unsigned short* Wmout = aB((size_t)DIN * HDIM);
  unsigned short* hfull = aB((size_t)BB * TT * HDIM);
  unsigned short* hev = aB((size_t)BB * EEV * HDIM);
  float* zx = aF((size_t)BB * EEV * ZDTD);
  float* xbc = aF((size_t)BB * EEV * CONVD);
  float* dtv = aF((size_t)BB * EEV * NHEAD);
  float* dec = aF((size_t)BB * EEV * NHEAD);
  float* ysc = aF((size_t)BB * EEV * DIN);
  unsigned short* memy = aB((size_t)BB * MMEM * DIN);
  float* mact = aF((size_t)BB * MMEM * HDIM);
  unsigned short* mactb = aB((size_t)BB * MMEM * HDIM);
  unsigned short* h2 = aB((size_t)BB * TT2 * HDIM);
  float* resid = aF((size_t)BB * TT2 * HDIM);
  float* c2 = aF((size_t)BB * TT2 * DHEAD);
  float* s2 = aF((size_t)BB * TT2 * DHEAD);
  float* qf = aF((size_t)BB * TT2 * NHEAD * DHEAD);
  float* kf = aF((size_t)BB * TT2 * KVHEAD * DHEAD);
  float* vf = aF((size_t)BB * TT2 * KVHEAD * DHEAD);
  float* mkf = aF((size_t)BB * MMEM * KVHEAD * DHEAD);
  float* mvf = aF((size_t)BB * MMEM * KVHEAD * DHEAD);
  unsigned short* qb = aB((size_t)BB * NHEAD * TT2 * DHEAD);
  unsigned short* kb = aB((size_t)BB * KVHEAD * TT2 * DHEAD);
  unsigned short* vb = aB((size_t)BB * KVHEAD * TT2 * DHEAD);
  unsigned short* ob = aB((size_t)BB * TT2 * NHEAD * DHEAD);
  float* yres = aF((size_t)BB * TT2 * HDIM);
  unsigned short* h3 = aB((size_t)BB * TT2 * HDIM);
  unsigned short* gate = aB((size_t)BB * TT2 * FFD);

  auto cvt = [&](const float* s, unsigned short* d, size_t n) {
    cvt_bf16<<<(unsigned)((n + 255) / 256), 256, 0, stream>>>(s, d, n);
  };
  // weight conversions (cached in L2 across GEMMs)
  cvt(wq, Wq, (size_t)HDIM * NHEAD * DHEAD);
  cvt(wk, Wk, (size_t)HDIM * KVHEAD * DHEAD);
  cvt(wv, Wv, (size_t)HDIM * KVHEAD * DHEAD);
  cvt(wo, Wo, (size_t)NHEAD * DHEAD * HDIM);
  cvt(wg, Wg, (size_t)HDIM * FFD);
  cvt(wu, Wu, (size_t)HDIM * FFD);
  cvt(wd, Wd, (size_t)FFD * HDIM);
  cvt(m_in_w, Wmin, (size_t)HDIM * ZDTD);
  cvt(m_out_w, Wmout, (size_t)DIN * HDIM);

  const bf16_t* bWq = (const bf16_t*)Wq;   const bf16_t* bWk = (const bf16_t*)Wk;
  const bf16_t* bWv = (const bf16_t*)Wv;   const bf16_t* bWo = (const bf16_t*)Wo;
  const bf16_t* bWg = (const bf16_t*)Wg;   const bf16_t* bWu = (const bf16_t*)Wu;
  const bf16_t* bWd = (const bf16_t*)Wd;   const bf16_t* bWmin = (const bf16_t*)Wmin;
  const bf16_t* bWmout = (const bf16_t*)Wmout;

  // ln1 over full sequence
  rmsnorm_bf16<<<BB * TT, 256, 0, stream>>>(x, ln1, hfull, HDIM);

  // ---- Mamba2 over evicted span ----
  gather_hev<<<(unsigned)(((size_t)BB * EEV * HDIM) / 256), 256, 0, stream>>>(hfull, hev);
  gemm_k<false, false><<<dim3((ZDTD + 127) / 128, (BB * EEV) / 64), 256, 0, stream>>>(
      (const bf16_t*)hev, bWmin, nullptr, nullptr, nullptr, zx, nullptr,
      BB * EEV, ZDTD, HDIM);
  mamba_conv<<<(unsigned)(((size_t)BB * EEV * CONVD) / 256), 256, 0, stream>>>(
      zx, m_conv_w, m_conv_b, xbc);
  dt_decay<<<(unsigned)(((size_t)BB * EEV * NHEAD) / 256), 256, 0, stream>>>(
      zx, m_dt_bias, m_A_log, dtv, dec);
  mamba_scan<<<BB * NHEAD, 256, 0, stream>>>(xbc, dtv, dec, ysc);
  mem_norm<<<BB * MMEM, 256, 0, stream>>>(ysc, zx, xbc, m_D, m_norm_w, midx, memy);
  gemm_k<false, false><<<dim3((HDIM + 127) / 128, (BB * MMEM) / 64), 256, 0, stream>>>(
      (const bf16_t*)memy, bWmout, nullptr, nullptr, nullptr, mact, nullptr,
      BB * MMEM, HDIM, DIN);
  cvt(mact, mactb, (size_t)BB * MMEM * HDIM);

  // ---- attention inputs ----
  gather_tokens<<<(unsigned)(((size_t)BB * TT2 * HDIM) / 256), 256, 0, stream>>>(
      hfull, x, h2, resid);
  gather_cs<<<(unsigned)(((size_t)BB * TT2 * DHEAD) / 256), 256, 0, stream>>>(
      cosA, sinA, c2, s2);
  gemm_k<false, false><<<dim3((NHEAD * DHEAD + 127) / 128, (BB * TT2) / 64), 256, 0,
                         stream>>>((const bf16_t*)h2, bWq, nullptr, bq, nullptr, qf,
                                   nullptr, BB * TT2, NHEAD * DHEAD, HDIM);
  gemm_k<false, false><<<dim3((KVHEAD * DHEAD + 127) / 128, (BB * TT2) / 64), 256, 0,
                         stream>>>((const bf16_t*)h2, bWk, nullptr, bk, nullptr, kf,
                                   nullptr, BB * TT2, KVHEAD * DHEAD, HDIM);
  gemm_k<false, false><<<dim3((KVHEAD * DHEAD + 127) / 128, (BB * TT2) / 64), 256, 0,
                         stream>>>((const bf16_t*)h2, bWv, nullptr, bv, nullptr, vf,
                                   nullptr, BB * TT2, KVHEAD * DHEAD, HDIM);
  gemm_k<false, false><<<dim3((KVHEAD * DHEAD + 127) / 128, (BB * MMEM) / 64), 256, 0,
                         stream>>>((const bf16_t*)mactb, bWk, nullptr, bk, nullptr,
                                   mkf, nullptr, BB * MMEM, KVHEAD * DHEAD, HDIM);
  gemm_k<false, false><<<dim3((KVHEAD * DHEAD + 127) / 128, (BB * MMEM) / 64), 256, 0,
                         stream>>>((const bf16_t*)mactb, bWv, nullptr, bv, nullptr,
                                   mvf, nullptr, BB * MMEM, KVHEAD * DHEAD, HDIM);
  rope_q<<<(unsigned)(((size_t)BB * NHEAD * TT2 * DHEAD) / 256), 256, 0, stream>>>(
      qf, c2, s2, qb);
  assemble_k<<<(unsigned)(((size_t)BB * KVHEAD * TT2 * DHEAD) / 256), 256, 0, stream>>>(
      kf, mkf, c2, s2, kb);
  assemble_v<<<(unsigned)(((size_t)BB * KVHEAD * TT2 * DHEAD) / 256), 256, 0, stream>>>(
      vf, mvf, vb);

  // ---- flash attention ----
  flash_attn<<<(BB * NHEAD * (TT2 / 16)) / 8, 256, 0, stream>>>(
      (const bf16_t*)qb, (const bf16_t*)kb, (const bf16_t*)vb, ob);

  // ---- output projection + residual ----
  gemm_k<false, false><<<dim3((HDIM + 127) / 128, (BB * TT2) / 64), 256, 0, stream>>>(
      (const bf16_t*)ob, bWo, nullptr, nullptr, resid, yres, nullptr, BB * TT2,
      HDIM, NHEAD * DHEAD);

  // ---- MLP ----
  rmsnorm_bf16<<<BB * TT2, 256, 0, stream>>>(yres, ln2, h3, HDIM);
  gemm_k<true, true><<<dim3((FFD + 127) / 128, (BB * TT2) / 64), 256, 0, stream>>>(
      (const bf16_t*)h3, bWg, bWu, nullptr, nullptr, nullptr, gate, BB * TT2, FFD,
      HDIM);
  gemm_k<false, false><<<dim3((HDIM + 127) / 128, (BB * TT2) / 64), 256, 0, stream>>>(
      (const bf16_t*)gate, bWd, nullptr, nullptr, yres, (float*)d_out, nullptr,
      BB * TT2, HDIM, FFD);
}
